// CausalSelfAttention_28673201668621
// MI455X (gfx1250) — matrix operand order repaired and checked by reading, not compiled, not run on hardware
//
#include <hip/hip_runtime.h>

// ---------------------------------------------------------------------------
// CDNA5 / gfx1250 wave32 WMMA implementation of windowed causal attention.
// All matmuls run on v_wmma_f32_16x16x32_bf16 (bf16 in, f32 accumulate).
// ---------------------------------------------------------------------------

typedef __attribute__((ext_vector_type(16))) __bf16 v16bf;
typedef __attribute__((ext_vector_type(8)))  __bf16 v8bf;
typedef __attribute__((ext_vector_type(8)))  float  v8f;
typedef __attribute__((ext_vector_type(4)))  float  v4f;

constexpr int B_  = 4;
constexpr int T_  = 2048;
constexpr int C_  = 768;
constexpr int NH_ = 12;
constexpr int HD_ = 64;
constexpr int M_  = B_ * T_;          // 8192 token rows

// ---- workspace layout (bytes) --------------------------------------------
constexpr size_t SZ_ACT = (size_t)M_ * C_ * 2;   // bf16 activation tensor
constexpr size_t SZ_W   = (size_t)C_ * C_ * 2;   // bf16 weight matrix
constexpr size_t OFF_XB  = 0;
constexpr size_t OFF_QB  = OFF_XB  + SZ_ACT;     // q bf16 [B,H,T,HD]
constexpr size_t OFF_KB  = OFF_QB  + SZ_ACT;     // k bf16 [B,H,T,HD]
constexpr size_t OFF_VTB = OFF_KB  + SZ_ACT;     // v bf16 [B,H,HD,T] (transposed)
constexpr size_t OFF_OB  = OFF_VTB + SZ_ACT;     // attn out bf16 [B,T,C]
constexpr size_t OFF_WQ  = OFF_OB  + SZ_ACT;
constexpr size_t OFF_WK  = OFF_WQ  + SZ_W;
constexpr size_t OFF_WV  = OFF_WK  + SZ_W;
constexpr size_t OFF_WP  = OFF_WV  + SZ_W;       // total ~67.6 MB

// ---- helpers --------------------------------------------------------------
__device__ inline __bf16 f2bf(float f) {
  unsigned u = __float_as_uint(f);
  unsigned r = (u + 0x7fffu + ((u >> 16) & 1u)) >> 16;
  unsigned short s = (unsigned short)r;
  __bf16 o;
  __builtin_memcpy(&o, &s, 2);
  return o;
}

__device__ inline v8f zero8() {
  v8f z;
#pragma unroll
  for (int i = 0; i < 8; ++i) z[i] = 0.f;
  return z;
}

// A-matrix fragment (16x32 bf16): lane = M (mod 16);
// lane-half selects K sub-block: elems j -> K = half*16 + j
// (matches B fragments).  row_ptr points at A[row][k0].
__device__ inline v16bf make_a(const __bf16* row_ptr, int half) {
  v8bf lo = *(const v8bf*)(row_ptr + half * 16);
  v8bf hi = *(const v8bf*)(row_ptr + half * 16 + 8);
  v16bf a;
#pragma unroll
  for (int i = 0; i < 8; ++i) { a[i] = lo[i]; a[i + 8] = hi[i]; }
  return a;
}

__device__ inline v8f wmma_bf16(v16bf a, v16bf b, v8f c) {
  return __builtin_amdgcn_wmma_f32_16x16x32_bf16(
      false, a, false, b, (short)0, c, false, false);
}

// reductions across the 16-lane half-wave (D-layout: one row spans 16 lanes)
__device__ inline float hmax16(float v) {
#pragma unroll
  for (int m = 1; m < 16; m <<= 1) v = fmaxf(v, __shfl_xor(v, m, 32));
  return v;
}
__device__ inline float hsum16(float v) {
#pragma unroll
  for (int m = 1; m < 16; m <<= 1) v += __shfl_xor(v, m, 32);
  return v;
}

// ---------------------------------------------------------------------------
// fp32 -> bf16 conversion, 8 elements per lane (2x b128 load, 1x b128 store)
// ---------------------------------------------------------------------------
__global__ void cvt_bf16_kernel(const float* __restrict__ src,
                                __bf16* __restrict__ dst, int n) {
  int i = (blockIdx.x * blockDim.x + threadIdx.x) * 8;
  int stride = gridDim.x * blockDim.x * 8;
  for (; i < n; i += stride) {
    v4f f0 = *(const v4f*)(src + i);
    v4f f1 = *(const v4f*)(src + i + 4);
    v8bf o;
#pragma unroll
    for (int j = 0; j < 4; ++j) { o[j] = f2bf(f0[j]); o[j + 4] = f2bf(f1[j]); }
    *(v8bf*)(dst + i) = o;
  }
}

// ---------------------------------------------------------------------------
// GEMM  out[m][n] = sum_k A[m][k] * W[n][k]   (A: [M_,C_], W: [C_,C_], bf16)
// block: 256 threads = 8 waves; each wave -> 32(M) x 64(N) tile (N tile = head),
// so each 32-wide K step does 8 WMMAs against 12 b128 loads.
// MODE 0: rotary+rmsnorm -> Q bf16 [B,H,T,HD]
// MODE 1: rotary+rmsnorm -> K bf16 [B,H,T,HD]
// MODE 2: + gate*ve      -> V bf16 [B,H,HD,T] (transposed)
// MODE 3: plain f32 store [M_,C_]  (output projection)
// ---------------------------------------------------------------------------
template <int MODE>
__global__ __launch_bounds__(256) void gemm_epi(
    const __bf16* __restrict__ A, const __bf16* __restrict__ W,
    __bf16* __restrict__ out_bf, float* __restrict__ out_f,
    const float* __restrict__ xf, const float* __restrict__ ve,
    const float* __restrict__ cosT, const float* __restrict__ sinT,
    const float* __restrict__ wgate) {
  const int lane = threadIdx.x & 31;
  const int wav  = threadIdx.x >> 5;
  const int half = lane >> 4;
  const int nlo  = lane & 15;
  const int mbase = blockIdx.x * 256 + wav * 32;   // 32 M-rows per wave
  const int h     = blockIdx.y;                    // 64-wide N tile == one head
  const int nbase = h * 64;

  v8f acc[2][4];
#pragma unroll
  for (int g = 0; g < 2; ++g)
#pragma unroll
    for (int i = 0; i < 4; ++i) acc[g][i] = zero8();

  const __bf16* arow0 = A + (size_t)(mbase + nlo) * C_;
  const __bf16* arow1 = A + (size_t)(mbase + 16 + nlo) * C_;

  for (int kk = 0; kk < C_; kk += 32) {
    v16bf a0 = make_a(arow0 + kk, half);
    v16bf a1 = make_a(arow1 + kk, half);
#pragma unroll
    for (int nt = 0; nt < 4; ++nt) {
      // B fragment (32x16): lane = N (mod 16), elems j -> K = half*16 + j
      const __bf16* bp =
          W + (size_t)(nbase + nt * 16 + nlo) * C_ + kk + half * 16;
      v16bf b = *(const v16bf*)bp;
      acc[0][nt] = wmma_bf16(a0, b, acc[0][nt]);
      acc[1][nt] = wmma_bf16(a1, b, acc[1][nt]);
    }
  }

#pragma unroll
  for (int g = 0; g < 2; ++g) {
    const int rbase = mbase + g * 16;
    if (MODE == 0 || MODE == 1) {
      // rotary: pairs (d, d+32) live in tiles (nt, nt+2) of the same lane
#pragma unroll
      for (int nt = 0; nt < 2; ++nt) {
        int d = nt * 16 + nlo;              // 0..31
#pragma unroll
        for (int v = 0; v < 8; ++v) {
          int row = rbase + v + 8 * half;
          int t = row & (T_ - 1);
          float c = cosT[t * 32 + d];
          float s = sinT[t * 32 + d];
          float x1 = acc[g][nt][v], x2 = acc[g][nt + 2][v];
          acc[g][nt][v]     = x1 * c - x2 * s;
          acc[g][nt + 2][v] = x1 * s + x2 * c;
        }
      }
      // rms_norm over the 64-wide head, then *1.2
#pragma unroll
      for (int v = 0; v < 8; ++v) {
        float ss = 0.f;
#pragma unroll
        for (int nt = 0; nt < 4; ++nt) ss += acc[g][nt][v] * acc[g][nt][v];
        ss = hsum16(ss);
        float r = rsqrtf(ss * (1.0f / HD_) + 1e-6f) * 1.2f;
#pragma unroll
        for (int nt = 0; nt < 4; ++nt) acc[g][nt][v] *= r;
      }
      // store [B,H,T,HD] bf16
#pragma unroll
      for (int nt = 0; nt < 4; ++nt) {
        int d = nt * 16 + nlo;
#pragma unroll
        for (int v = 0; v < 8; ++v) {
          int row = rbase + v + 8 * half;
          int t = row & (T_ - 1);
          int b = row >> 11;
          out_bf[(((size_t)(b * NH_ + h) * T_ + t) * HD_) + d] =
              f2bf(acc[g][nt][v]);
        }
      }
    } else if (MODE == 2) {
#pragma unroll
      for (int v = 0; v < 8; ++v) {
        int row = rbase + v + 8 * half;
        int t = row & (T_ - 1);
        int b = row >> 11;
        float gg = 0.f;
#pragma unroll
        for (int j = 0; j < 12; ++j)
          gg += xf[(size_t)row * C_ + j] * wgate[h * 12 + j];
        gg = 3.0f / (1.0f + __expf(-gg));
#pragma unroll
        for (int nt = 0; nt < 4; ++nt) {
          int d = nt * 16 + nlo;
          float val = acc[g][nt][v] + gg * ve[(size_t)row * C_ + h * HD_ + d];
          // transposed store: [B,H,HD,T]
          out_bf[(((size_t)(b * NH_ + h) * HD_ + d) * T_) + t] = f2bf(val);
        }
      }
    } else {  // MODE 3: plain f32 store
#pragma unroll
      for (int nt = 0; nt < 4; ++nt) {
#pragma unroll
        for (int v = 0; v < 8; ++v) {
          int row = rbase + v + 8 * half;
          out_f[(size_t)row * C_ + nbase + nt * 16 + nlo] = acc[g][nt][v];
        }
      }
    }
  }
}

// ---------------------------------------------------------------------------
// Flash attention with sliding window.
// grid (T/64, NH, B), block 128 = 4 waves; each wave owns 16 queries.
// ---------------------------------------------------------------------------
__global__ __launch_bounds__(128) void attn_kernel(
    const __bf16* __restrict__ qb, const __bf16* __restrict__ kb,
    const __bf16* __restrict__ vtb, __bf16* __restrict__ ob,
    const int* __restrict__ wlp) {
  __shared__ __align__(16) __bf16 ldsP[4][16][32];   // per-wave P staging

  const int wl   = *wlp;
  const int lane = threadIdx.x & 31;
  const int w    = threadIdx.x >> 5;
  const int half = lane >> 4;
  const int nlo  = lane & 15;
  const int h = blockIdx.y, b = blockIdx.z;
  const int qbase = blockIdx.x * 64 + w * 16;

  const __bf16* qp = qb  + (size_t)(b * NH_ + h) * T_ * HD_;
  const __bf16* kp = kb  + (size_t)(b * NH_ + h) * T_ * HD_;
  const __bf16* vp = vtb + (size_t)(b * NH_ + h) * HD_ * T_;

  // Q fragments for the 16 rows of this wave (hd 0..31 / 32..63)
  v16bf aq[2];
  {
    const __bf16* rp = qp + (size_t)(qbase + nlo) * HD_;
    aq[0] = make_a(rp, half);
    aq[1] = make_a(rp + 32, half);
  }

  v8f o[4];
#pragma unroll
  for (int i = 0; i < 4; ++i) o[i] = zero8();
  float m[8], l[8];
#pragma unroll
  for (int v = 0; v < 8; ++v) { m[v] = -3.0e38f; l[v] = 0.f; }

  const float NEG_INF = -__builtin_huge_valf();
  int kbeg = qbase - wl;
  if (kbeg < 0) kbeg = 0;
  kbeg &= ~31;
  const int kend = qbase + 16;

  for (int kc = kbeg; kc < kend; kc += 32) {
    // ---- S = Q @ K^T for a 16x32 key slab (K-reduction over hd in 2 steps)
    v8f s0 = zero8(), s1 = zero8();
#pragma unroll
    for (int c = 0; c < 2; ++c) {
      v16bf bk0 = *(const v16bf*)(kp + (size_t)(kc + nlo) * HD_ + c * 32 + half * 16);
      v16bf bk1 = *(const v16bf*)(kp + (size_t)(kc + 16 + nlo) * HD_ + c * 32 + half * 16);
      s0 = wmma_bf16(aq[c], bk0, s0);
      s1 = wmma_bf16(aq[c], bk1, s1);
    }

    // ---- mask + online softmax update (D-layout: row spans 16 lanes)
    float p0[8], p1[8], alpha[8];
#pragma unroll
    for (int v = 0; v < 8; ++v) {
      int row = qbase + v + 8 * half;
      int c0 = kc + nlo, c1 = kc + 16 + nlo;
      float sv0 = (c0 <= row && row - c0 <= wl) ? s0[v] * 0.125f : NEG_INF;
      float sv1 = (c1 <= row && row - c1 <= wl) ? s1[v] * 0.125f : NEG_INF;
      float mx = hmax16(fmaxf(sv0, sv1));
      float mn = fmaxf(m[v], mx);
      alpha[v] = __expf(m[v] - mn);
      m[v] = mn;
      p0[v] = __expf(sv0 - mn);        // exp(-inf) == 0 for masked cols
      p1[v] = __expf(sv1 - mn);
      l[v] = l[v] * alpha[v] + hsum16(p0[v] + p1[v]);
    }
#pragma unroll
    for (int nt = 0; nt < 4; ++nt)
#pragma unroll
      for (int v = 0; v < 8; ++v) o[nt][v] *= alpha[v];

    // ---- stage P through LDS: D-layout -> A-layout
#pragma unroll
    for (int v = 0; v < 8; ++v) {
      ldsP[w][v + 8 * half][nlo]      = f2bf(p0[v]);
      ldsP[w][v + 8 * half][16 + nlo] = f2bf(p1[v]);
    }
    v16bf pa = make_a(&ldsP[w][nlo][0], half);   // same-wave RAW: HW waits DScnt

    // ---- O += P @ V  (V transposed: B-operand contiguous along keys)
#pragma unroll
    for (int nt = 0; nt < 4; ++nt) {
      int hd = nt * 16 + nlo;
      v16bf bv = *(const v16bf*)(vp + (size_t)hd * T_ + kc + half * 16);
      o[nt] = wmma_bf16(pa, bv, o[nt]);
    }
  }

  // ---- normalize and store bf16 [B,T,C] (head-interleaved)
  float inv[8];
#pragma unroll
  for (int v = 0; v < 8; ++v) inv[v] = 1.0f / l[v];
#pragma unroll
  for (int nt = 0; nt < 4; ++nt) {
#pragma unroll
    for (int v = 0; v < 8; ++v) {
      int t = qbase + v + 8 * half;
      int col = h * HD_ + nt * 16 + nlo;
      ob[((size_t)(b * T_ + t)) * C_ + col] = f2bf(o[nt][v] * inv[v]);
    }
  }
}

// ---------------------------------------------------------------------------
extern "C" void kernel_launch(void* const* d_in, const int* in_sizes, int n_in,
                              void* d_out, int out_size, void* d_ws,
                              size_t ws_size, hipStream_t stream) {
  (void)in_sizes; (void)n_in; (void)out_size; (void)ws_size;
  const float* x     = (const float*)d_in[0];
  const float* ve    = (const float*)d_in[1];
  const float* cosT  = (const float*)d_in[2];
  const float* sinT  = (const float*)d_in[3];
  const float* Wq    = (const float*)d_in[4];
  const float* Wk    = (const float*)d_in[5];
  const float* Wv    = (const float*)d_in[6];
  const float* Wproj = (const float*)d_in[7];
  const float* Wg    = (const float*)d_in[8];
  const int*   wlp   = (const int*)d_in[9];
  float* out = (float*)d_out;
  char*  ws  = (char*)d_ws;

  __bf16* xb  = (__bf16*)(ws + OFF_XB);
  __bf16* qb  = (__bf16*)(ws + OFF_QB);
  __bf16* kb  = (__bf16*)(ws + OFF_KB);
  __bf16* vtb = (__bf16*)(ws + OFF_VTB);
  __bf16* ob  = (__bf16*)(ws + OFF_OB);
  __bf16* wqb = (__bf16*)(ws + OFF_WQ);
  __bf16* wkb = (__bf16*)(ws + OFF_WK);
  __bf16* wvb = (__bf16*)(ws + OFF_WV);
  __bf16* wpb = (__bf16*)(ws + OFF_WP);

  // fp32 -> bf16 conversions (8 elems per lane)
  cvt_bf16_kernel<<<1024, 256, 0, stream>>>(x, xb, M_ * C_);
  cvt_bf16_kernel<<<288, 256, 0, stream>>>(Wq, wqb, C_ * C_);
  cvt_bf16_kernel<<<288, 256, 0, stream>>>(Wk, wkb, C_ * C_);
  cvt_bf16_kernel<<<288, 256, 0, stream>>>(Wv, wvb, C_ * C_);
  cvt_bf16_kernel<<<288, 256, 0, stream>>>(Wproj, wpb, C_ * C_);

  dim3 gg(M_ / 256, C_ / 64);   // (32, 12)
  gemm_epi<0><<<gg, 256, 0, stream>>>(xb, wqb, qb,  nullptr, x, ve, cosT, sinT, Wg);
  gemm_epi<1><<<gg, 256, 0, stream>>>(xb, wkb, kb,  nullptr, x, ve, cosT, sinT, Wg);
  gemm_epi<2><<<gg, 256, 0, stream>>>(xb, wvb, vtb, nullptr, x, ve, cosT, sinT, Wg);

  attn_kernel<<<dim3(T_ / 64, NH_, B_), 128, 0, stream>>>(qb, kb, vtb, ob, wlp);

  gemm_epi<3><<<gg, 256, 0, stream>>>(ob, wpb, nullptr, out, x, ve, cosT, sinT, Wg);
}
